// RouteGeneratorModel_6717328851336
// MI455X (gfx1250) — compile-verified, matrix operand order checked by reading, not run on hardware
//
#include <hip/hip_runtime.h>
#include <hip/hip_bf16.h>
#include <math.h>

// ---------------------------------------------------------------------------
// MI455X (gfx1250, wave32) implementation.
// Pipeline:
//   1. convert x -> bf16; transpose+convert cw1, sw1, sw2 -> bf16 [N,K]
//   2. WMMA bf16 GEMM: h_c = relu(x@cw1+cb1)  (f32 out)
//   3. WMMA bf16 GEMM: h_s = relu(x@sw1+sb1)  (bf16 out, feeds next GEMM)
//   4. WMMA bf16 GEMM: scores = h_s@sw2+sb2   (f32, 4096x50000, in ws)
//      -> B panel staged in LDS via TENSOR_LOAD_TO_LDS (TDM), shared by all
//         8 waves of the block; coalesced float4 writeback via LDS staging.
//   5. complexity = sigmoid(h_c . cw2 + cb2)
//   6. per-row 2048-bin radix-select top-`budget` + softmax
// budget recovered on host from out_size (deterministic, graph-safe).
// ---------------------------------------------------------------------------

typedef __attribute__((ext_vector_type(16))) __bf16 v16bf;
typedef __attribute__((ext_vector_type(8)))  __bf16 v8bf;
typedef __attribute__((ext_vector_type(8)))  float  v8f;
typedef __attribute__((ext_vector_type(4)))  unsigned int v4u;
typedef __attribute__((ext_vector_type(8)))  int v8i;
typedef __attribute__((ext_vector_type(4)))  int v4i;

#define B_ROWS 4096
#define DIM    512
#define POOL   50000
#define H_C    128
#define H_S    256

#if defined(__has_builtin)
#  if __has_builtin(__builtin_amdgcn_tensor_load_to_lds)
#    define HAVE_TDM 1
#  endif
#endif
#ifndef HAVE_TDM
#  define HAVE_TDM 0
#endif

// ---- WMMA fragment loaders (16-bit A/B layout, ISA 7.12.2) -----------------
// Operand is row-major [16, K] with row stride `ld` (elements), tile K offset k0.
// lane 0-15: row = lane, holds K = k0+{0..7} and k0+16+{0..7}
// lane 16-31: row = lane-16, holds K = k0+8+{0..7} and k0+24+{0..7}
__device__ __forceinline__ v16bf load_frag16(const __bf16* __restrict__ p, int ld, int k0) {
    int lane = threadIdx.x & 31;
    int row  = lane & 15;
    int kb   = (lane >> 4) << 3;              // 0 or 8
    union { v16bf v; v8bf h[2]; } u;
    u.h[0] = *(const v8bf*)(p + (size_t)row * ld + k0 + kb);
    u.h[1] = *(const v8bf*)(p + (size_t)row * ld + k0 + 16 + kb);
    return u.v;
}

__device__ __forceinline__ v8f wmma_bf16(v16bf a, v16bf b, v8f c) {
    return __builtin_amdgcn_wmma_f32_16x16x32_bf16(
        /*neg_a=*/false, a, /*neg_b=*/false, b,
        /*c_mod=*/(short)0, c, /*reuse_a=*/false, /*reuse_b=*/false);
}

// ---- conversion / transpose kernels ---------------------------------------
__global__ void k_f32_to_bf16(const float* __restrict__ in, __bf16* __restrict__ out, int n) {
    int i = blockIdx.x * blockDim.x + threadIdx.x;
    if (i < n) out[i] = (__bf16)in[i];
}

// in: [K,N] f32 row-major  ->  out: [N,K] bf16 row-major (LDS-tiled)
__global__ void k_transpose_bf16(const float* __restrict__ in, __bf16* __restrict__ out,
                                 int K, int N) {
    __shared__ float tile[32][33];
    int k0 = blockIdx.y * 32, n0 = blockIdx.x * 32;
    int tx = threadIdx.x, ty = threadIdx.y;     // 32 x 8
    for (int r = ty; r < 32; r += 8) {
        int k = k0 + r, n = n0 + tx;
        tile[r][tx] = (k < K && n < N) ? in[(size_t)k * N + n] : 0.0f;
    }
    __syncthreads();
    for (int r = ty; r < 32; r += 8) {
        int n = n0 + r, k = k0 + tx;
        if (n < N && k < K) out[(size_t)n * K + k] = (__bf16)tile[tx][r];
    }
}

// ---- GEMM: C = relu(A[M,K] @ W^T[N,K]^T + bias), bf16 output --------------
__global__ void k_gemm_relu_bf16out(const __bf16* __restrict__ A,
                                    const __bf16* __restrict__ WT,
                                    const float* __restrict__ bias,
                                    __bf16* __restrict__ H,
                                    int M, int N, int K) {
    int wave   = (blockIdx.x * blockDim.x + threadIdx.x) >> 5;
    int ntiles = N >> 4;
    int mt = wave / ntiles, nt = wave - mt * ntiles;
    if (mt >= (M >> 4)) return;
    const __bf16* Ap = A  + (size_t)(mt << 4) * K;
    const __bf16* Bp = WT + (size_t)(nt << 4) * K;
    v8f acc = {};
    #pragma unroll 4
    for (int k0 = 0; k0 < K; k0 += 32) {
        v16bf a = load_frag16(Ap, K, k0);
        v16bf b = load_frag16(Bp, K, k0);
        acc = wmma_bf16(a, b, acc);
    }
    int lane  = threadIdx.x & 31;
    int n     = (nt << 4) + (lane & 15);
    int mbase = (mt << 4) + ((lane >> 4) << 3);
    float bn  = bias[n];
    #pragma unroll
    for (int r = 0; r < 8; ++r) {
        float v = acc[r] + bn;
        v = v > 0.0f ? v : 0.0f;
        H[(size_t)(mbase + r) * N + n] = (__bf16)v;
    }
}

// ---- GEMM: C = relu(A @ W^T^T + bias), f32 output (complexity hidden) -----
__global__ void k_gemm_relu_f32out(const __bf16* __restrict__ A,
                                   const __bf16* __restrict__ WT,
                                   const float* __restrict__ bias,
                                   float* __restrict__ H,
                                   int M, int N, int K) {
    int wave   = (blockIdx.x * blockDim.x + threadIdx.x) >> 5;
    int ntiles = N >> 4;
    int mt = wave / ntiles, nt = wave - mt * ntiles;
    if (mt >= (M >> 4)) return;
    const __bf16* Ap = A  + (size_t)(mt << 4) * K;
    const __bf16* Bp = WT + (size_t)(nt << 4) * K;
    v8f acc = {};
    #pragma unroll 4
    for (int k0 = 0; k0 < K; k0 += 32) {
        v16bf a = load_frag16(Ap, K, k0);
        v16bf b = load_frag16(Bp, K, k0);
        acc = wmma_bf16(a, b, acc);
    }
    int lane  = threadIdx.x & 31;
    int n     = (nt << 4) + (lane & 15);
    int mbase = (mt << 4) + ((lane >> 4) << 3);
    float bn  = bias[n];
    #pragma unroll
    for (int r = 0; r < 8; ++r) {
        float v = acc[r] + bn;
        H[(size_t)(mbase + r) * N + n] = v > 0.0f ? v : 0.0f;
    }
}

// ---- big GEMM: scores[M,N] = Hs[M,K] @ sw2T[N,K]^T + sb2 ------------------
// Block = 256 threads = 8 waves. Block covers 128 rows (8 M-tiles, one/wave)
// x 80 cols (5 N-tiles). The 80x256 bf16 B panel (40 KB) is staged once in
// LDS by the Tensor Data Mover and shared by all 8 waves (8x cut in B
// traffic). Output tile staged in LDS for coalesced float4 writeback.
#define NT_BLK  5
#define NCOLS   (16 * NT_BLK)        /* 80 */
#define MB_ROWS 128                  /* 8 waves * 16 rows */
#define NBLKS   (POOL / NCOLS)       /* 625 */

__global__ void __launch_bounds__(256) k_gemm_scores(
        const __bf16* __restrict__ Hs,
        const __bf16* __restrict__ W2T,
        const float* __restrict__ sb2,
        float* __restrict__ scores) {
    __shared__ __bf16 Bpanel[NCOLS * H_S];            // 40 KB
    __shared__ float  Ostage[8 * 16 * NCOLS];         // 40 KB

    int nb   = blockIdx.x % NBLKS;
    int mb   = blockIdx.x / NBLKS;
    int wid  = threadIdx.x >> 5;
    int lane = threadIdx.x & 31;
    const __bf16* Bsrc = W2T + (size_t)(nb * NCOLS) * H_S;

    // ---- stage B panel global -> LDS --------------------------------------
#if HAVE_TDM
    if (threadIdx.x < 32) {
        unsigned long long ga = (unsigned long long)(size_t)Bsrc;
        unsigned ldsa = (unsigned)(size_t)(__bf16*)Bpanel;   // low 32b = LDS offset
        // D# group 0: count=1 | lds_addr | global_addr[56:0] | type=2
        v4u g0 = { 1u, ldsa, (unsigned)ga,
                   (unsigned)((ga >> 32) & 0x01FFFFFFu) | 0x80000000u };
        // D# group 1: data_size=2B; tensor 256x80; tile 256x80; dim0 stride 256
        v8i g1 = { (int)(1u << 16),            // workgroup_mask=0, data_size=1(2B)
                   (int)((unsigned)H_S << 16), // tensor_dim0[15:0] @ bits 63:48
                   (int)((unsigned)NCOLS << 16), // tensor_dim1[15:0] @ bits 95:80
                   (int)((unsigned)H_S << 16), // tile_dim0 @ bits 127:112
                   NCOLS,                      // tile_dim1 @ bits 143:128
                   H_S,                        // tensor_dim0_stride[31:0]
                   0, 0 };
        v4i gz = { 0, 0, 0, 0 };
#  if defined(__clang_major__) && __clang_major__ >= 23
        v8i gz8 = { 0, 0, 0, 0, 0, 0, 0, 0 };
        __builtin_amdgcn_tensor_load_to_lds(g0, g1, gz, gz, gz8, 0);
#  else
        __builtin_amdgcn_tensor_load_to_lds(g0, g1, gz, gz, 0);
#  endif
        __builtin_amdgcn_s_wait_tensorcnt(0);
    }
#else
    // fallback: panel is one contiguous 40960B block -> coalesced v8bf copy
    for (int i = threadIdx.x; i < (NCOLS * H_S) / 8; i += (int)blockDim.x)
        ((v8bf*)Bpanel)[i] = ((const v8bf*)Bsrc)[i];
#endif
    __syncthreads();

    // ---- WMMA main loop: A from global, B from LDS ------------------------
    const __bf16* Ap = Hs + (size_t)(mb * MB_ROWS + wid * 16) * H_S;
    v8f acc[NT_BLK];
    #pragma unroll
    for (int t = 0; t < NT_BLK; ++t) acc[t] = (v8f){};
    #pragma unroll
    for (int k0 = 0; k0 < H_S; k0 += 32) {
        __builtin_prefetch(Ap + k0 + 64, 0, 1);       // global_prefetch_b8
        v16bf a = load_frag16(Ap, H_S, k0);
        #pragma unroll
        for (int t = 0; t < NT_BLK; ++t) {
            v16bf b = load_frag16(Bpanel + (size_t)(t * 16) * H_S, H_S, k0);
            acc[t] = wmma_bf16(a, b, acc[t]);
        }
    }

    // ---- stage (acc + bias) to LDS, then coalesced float4 writeback -------
    float* ost = Ostage + wid * 16 * NCOLS;
    int col0  = lane & 15;
    int rbase = (lane >> 4) << 3;
    #pragma unroll
    for (int t = 0; t < NT_BLK; ++t) {
        int ncol = t * 16 + col0;
        float bn = sb2[nb * NCOLS + ncol];
        #pragma unroll
        for (int r = 0; r < 8; ++r)
            ost[(rbase + r) * NCOLS + ncol] = acc[t][r] + bn;
    }
    // same-wave DS ordering; compiler inserts the dscnt wait for the RAW dep
    const float4* osv = (const float4*)ost;
    size_t gbase = (size_t)(mb * MB_ROWS + wid * 16) * POOL + (size_t)nb * NCOLS;
    #pragma unroll
    for (int j = 0; j < (16 * NCOLS) / (32 * 4); ++j) {     // 10 float4 / lane
        int idx  = j * 32 + lane;
        int flat = idx * 4;
        int row  = flat / NCOLS;
        int colf = flat - row * NCOLS;
        *(float4*)(scores + gbase + (size_t)row * POOL + colf) = osv[idx];
    }
}

// ---- complexity: sigmoid(h_c . cw2 + cb2) ---------------------------------
__global__ void k_complexity(const float* __restrict__ Hc,
                             const float* __restrict__ cw2,
                             const float* __restrict__ cb2,
                             float* __restrict__ comp) {
    int row = blockIdx.x;
    int t   = threadIdx.x;                       // 128 threads
    float v = Hc[(size_t)row * H_C + t] * cw2[t];
    #pragma unroll
    for (int off = 16; off > 0; off >>= 1) v += __shfl_xor(v, off, 32);
    __shared__ float red[4];
    if ((t & 31) == 0) red[t >> 5] = v;
    __syncthreads();
    if (t == 0) {
        float s = red[0] + red[1] + red[2] + red[3] + cb2[0];
        comp[row] = 1.0f / (1.0f + __expf(-s));
    }
}

__global__ void k_write_budget(int* __restrict__ out, int budget) {
    if (threadIdx.x == 0 && blockIdx.x == 0) *out = budget;
}

// ---- per-row top-k (2048-bin radix select) + softmax ----------------------
#define BINS 2048
__global__ void k_topk_softmax(const float* __restrict__ scores,
                               int* __restrict__ idx_out,
                               float* __restrict__ w_out,
                               int budget, int N) {
    int row = blockIdx.x;
    const float* s = scores + (size_t)row * N;
    __shared__ unsigned hist[BINS];
    __shared__ float red[16];
    __shared__ int sT, sGt;
    __shared__ unsigned cHi, cTie;
    __shared__ float sMax, sSum;
    int nwaves = blockDim.x >> 5;

    for (int i = threadIdx.x; i < BINS; i += blockDim.x) hist[i] = 0;
    if (threadIdx.x == 0) { cHi = 0; cTie = 0; }
    __syncthreads();

    // pass 1: histogram of order-preserving key top bits + row max
    float lmax = -INFINITY;
    for (int i = threadIdx.x; i < N; i += blockDim.x) {
        float f = s[i];
        lmax = fmaxf(lmax, f);
        unsigned u = __float_as_uint(f);
        unsigned key = (u & 0x80000000u) ? ~u : (u | 0x80000000u);
        atomicAdd(&hist[key >> 21], 1u);
    }
    #pragma unroll
    for (int off = 16; off > 0; off >>= 1) lmax = fmaxf(lmax, __shfl_xor(lmax, off, 32));
    if ((threadIdx.x & 31) == 0) red[threadIdx.x >> 5] = lmax;
    __syncthreads();

    if (threadIdx.x == 0) {
        float m = red[0];
        for (int w = 1; w < nwaves; ++w) m = fmaxf(m, red[w]);
        sMax = m;
        unsigned cum = 0; int T = 0; unsigned gt = 0;
        for (int b = BINS - 1; b >= 0; --b) {
            unsigned c = hist[b];
            if (cum + c >= (unsigned)budget) { T = b; gt = cum; break; }
            cum += c;
            if (b == 0) { T = 0; gt = cum; }
        }
        sT = T; sGt = (int)gt;
    }
    __syncthreads();
    int T = sT, gt = sGt;
    size_t obase = (size_t)row * budget;

    // pass 2: compact winners (definite > T first, ties in bin T fill the rest)
    for (int i = threadIdx.x; i < N; i += blockDim.x) {
        float f = s[i];
        unsigned u = __float_as_uint(f);
        unsigned key = (u & 0x80000000u) ? ~u : (u | 0x80000000u);
        int b = (int)(key >> 21);
        if (b > T) {
            unsigned p = atomicAdd(&cHi, 1u);
            idx_out[obase + p] = i;
            w_out[obase + p]   = f;
        } else if (b == T) {
            unsigned p = atomicAdd(&cTie, 1u);
            if ((int)p < budget - gt) {
                idx_out[obase + gt + p] = i;
                w_out[obase + gt + p]   = f;
            }
        }
    }
    __syncthreads();

    // pass 3: softmax over selected scores (max == row max since top-1 kept)
    float m = sMax, lsum = 0.0f;
    for (int j = threadIdx.x; j < budget; j += blockDim.x)
        lsum += __expf(w_out[obase + j] - m);
    #pragma unroll
    for (int off = 16; off > 0; off >>= 1) lsum += __shfl_xor(lsum, off, 32);
    if ((threadIdx.x & 31) == 0) red[threadIdx.x >> 5] = lsum;
    __syncthreads();
    if (threadIdx.x == 0) {
        float t = 0.0f;
        for (int w = 0; w < nwaves; ++w) t += red[w];
        sSum = t;
    }
    __syncthreads();
    float inv = 1.0f / sSum;
    for (int j = threadIdx.x; j < budget; j += blockDim.x)
        w_out[obase + j] = __expf(w_out[obase + j] - m) * inv;
}

// ---------------------------------------------------------------------------
extern "C" void kernel_launch(void* const* d_in, const int* in_sizes, int n_in,
                              void* d_out, int out_size, void* d_ws, size_t ws_size,
                              hipStream_t stream) {
    const float* x   = (const float*)d_in[0];   // [4096,512]
    const float* cw1 = (const float*)d_in[1];   // [512,128]
    const float* cb1 = (const float*)d_in[2];   // [128]
    const float* cw2 = (const float*)d_in[3];   // [128,1]
    const float* cb2 = (const float*)d_in[4];   // [1]
    const float* sw1 = (const float*)d_in[5];   // [512,256]
    const float* sb1 = (const float*)d_in[6];   // [256]
    const float* sw2 = (const float*)d_in[7];   // [256,50000]
    const float* sb2 = (const float*)d_in[8];   // [50000]
    (void)in_sizes; (void)n_in; (void)ws_size;

    // budget recovered from the harness-sized output buffer:
    // out = indices[B*budget] ++ budget[1] ++ weights[B*budget] ++ complexity[B]
    long long budget = ((long long)out_size - 1 - B_ROWS) / (2LL * B_ROWS);
    if (budget < 1) budget = 1;

    // workspace carve-up (256B aligned)
    size_t off = 0;
    auto carve = [&](size_t bytes) { size_t o = off; off += (bytes + 255) & ~(size_t)255; return o; };
    __bf16* xb    = (__bf16*)((char*)d_ws + carve((size_t)B_ROWS * DIM * 2));
    __bf16* cw1T  = (__bf16*)((char*)d_ws + carve((size_t)H_C * DIM * 2));
    __bf16* sw1T  = (__bf16*)((char*)d_ws + carve((size_t)H_S * DIM * 2));
    __bf16* sw2T  = (__bf16*)((char*)d_ws + carve((size_t)POOL * H_S * 2));
    __bf16* hs    = (__bf16*)((char*)d_ws + carve((size_t)B_ROWS * H_S * 2));
    float*  hc    = (float*)((char*)d_ws + carve((size_t)B_ROWS * H_C * 4));
    float*  sc    = (float*)((char*)d_ws + carve((size_t)B_ROWS * POOL * 4));

    int*   idx_out  = (int*)d_out;
    int*   bud_out  = (int*)d_out + (size_t)B_ROWS * budget;
    float* w_out    = (float*)d_out + (size_t)B_ROWS * budget + 1;
    float* comp_out = (float*)d_out + 2 * (size_t)B_ROWS * budget + 1;

    // 1) conversions
    {
        int n = B_ROWS * DIM;
        k_f32_to_bf16<<<(n + 255) / 256, 256, 0, stream>>>(x, xb, n);
    }
    {
        dim3 blk(32, 8);
        dim3 g1((H_C + 31) / 32, (DIM + 31) / 32);
        k_transpose_bf16<<<g1, blk, 0, stream>>>(cw1, cw1T, DIM, H_C);
        dim3 g2((H_S + 31) / 32, (DIM + 31) / 32);
        k_transpose_bf16<<<g2, blk, 0, stream>>>(sw1, sw1T, DIM, H_S);
        dim3 g3((POOL + 31) / 32, (H_S + 31) / 32);
        k_transpose_bf16<<<g3, blk, 0, stream>>>(sw2, sw2T, H_S, POOL);
    }

    // 2) complexity hidden: [4096,128] = relu(x @ cw1 + cb1)
    {
        int waves = (B_ROWS / 16) * (H_C / 16);          // 2048
        k_gemm_relu_f32out<<<waves / 8, 256, 0, stream>>>(xb, cw1T, cb1, hc,
                                                          B_ROWS, H_C, DIM);
    }
    // 3) scorer hidden: [4096,256] = relu(x @ sw1 + sb1), bf16
    {
        int waves = (B_ROWS / 16) * (H_S / 16);          // 4096
        k_gemm_relu_bf16out<<<waves / 8, 256, 0, stream>>>(xb, sw1T, sb1, hs,
                                                           B_ROWS, H_S, DIM);
    }
    // 4) scores: [4096,50000] = hs @ sw2 + sb2 (TDM-staged B panel in LDS)
    {
        int blocks = (B_ROWS / MB_ROWS) * NBLKS;         // 32*625 = 20000
        k_gemm_scores<<<blocks, 256, 0, stream>>>(hs, sw2T, sb2, sc);
    }
    // 5) complexity + budget outputs
    k_complexity<<<B_ROWS, H_C, 0, stream>>>(hc, cw2, cb2, comp_out);
    k_write_budget<<<1, 1, 0, stream>>>(bud_out, (int)budget);

    // 6) per-row top-k + softmax
    k_topk_softmax<<<B_ROWS, 512, 0, stream>>>(sc, idx_out, w_out, (int)budget, POOL);
}
